// DenseBlock_MHSA_32615981646444
// MI455X (gfx1250) — matrix operand order repaired and checked by reading, not compiled, not run on hardware
//
#include <hip/hip_runtime.h>

// ---------------------------------------------------------------------------
// DenseBlock + MHSA for MI455X (gfx1250, wave32, WMMA bf16)
// B=4096, C=F=512, H=W=3  -> T = 36864 tokens, 512 channels per tensor.
// ---------------------------------------------------------------------------

#define TOK   36864          // 4096 * 9
#define NB    4096
#define CCH   512
#define HWN   9
#define NTOT  (NB * CCH * HWN)   // 18,874,368
#define EPSBN 1e-5f

typedef __attribute__((ext_vector_type(16))) __bf16 v16bf;
typedef __attribute__((ext_vector_type(8)))  float  v8f;

__device__ __forceinline__ unsigned short f2bf(float f) {
    unsigned u = __float_as_uint(f);
    u += 0x7FFFu + ((u >> 16) & 1u);          // round-to-nearest-even
    return (unsigned short)(u >> 16);
}

// ---------------- elementwise converts ----------------
__global__ void k_f2bf(const float* __restrict__ src, unsigned short* __restrict__ dst, int n) {
    for (int i = blockIdx.x * blockDim.x + threadIdx.x; i < n; i += gridDim.x * blockDim.x)
        dst[i] = f2bf(src[i]);
}

// ---------------- channel statistics (deterministic two-stage) ----------------
// x in NCHW [4096,512,3,3]; partial layout: [P][ sum[512] | sumsq[512] ]
__global__ __launch_bounds__(256) void k_stats_nchw_partial(const float* __restrict__ x,
                                                            float* __restrict__ partial) {
    const int p = blockIdx.x;            // 128 blocks, 32 batch items each
    const int c0 = threadIdx.x, c1 = threadIdx.x + 256;
    float s0 = 0.f, q0 = 0.f, s1 = 0.f, q1 = 0.f;
    for (int b = p * 32; b < p * 32 + 32; ++b) {
        const float* r0 = x + ((size_t)b * CCH + c0) * HWN;
        const float* r1 = x + ((size_t)b * CCH + c1) * HWN;
        #pragma unroll
        for (int hw = 0; hw < HWN; ++hw) {
            float v = r0[hw]; s0 += v; q0 += v * v;
            v = r1[hw];       s1 += v; q1 += v * v;
        }
    }
    partial[p * 1024 + c0]       = s0;
    partial[p * 1024 + c1]       = s1;
    partial[p * 1024 + 512 + c0] = q0;
    partial[p * 1024 + 512 + c1] = q1;
}

// token-major [T,512] source
__global__ __launch_bounds__(256) void k_stats_tok_partial(const float* __restrict__ src,
                                                           float* __restrict__ partial) {
    const int p = blockIdx.x;            // 144 blocks, 256 rows each
    const int c0 = threadIdx.x, c1 = threadIdx.x + 256;
    float s0 = 0.f, q0 = 0.f, s1 = 0.f, q1 = 0.f;
    for (int r = p * 256; r < p * 256 + 256; ++r) {
        float v = src[(size_t)r * CCH + c0]; s0 += v; q0 += v * v;
        v       = src[(size_t)r * CCH + c1]; s1 += v; q1 += v * v;
    }
    partial[p * 1024 + c0]       = s0;
    partial[p * 1024 + c1]       = s1;
    partial[p * 1024 + 512 + c0] = q0;
    partial[p * 1024 + 512 + c1] = q1;
}

__global__ __launch_bounds__(512) void k_stats_finalize(const float* __restrict__ partial, int P,
                                                        float invN, float* __restrict__ mean,
                                                        float* __restrict__ rstd) {
    const int c = threadIdx.x;
    float s = 0.f, q = 0.f;
    for (int p = 0; p < P; ++p) {
        s += partial[p * 1024 + c];
        q += partial[p * 1024 + 512 + c];
    }
    const float m = s * invN;
    const float var = q * invN - m * m;       // biased variance (ddof=0)
    mean[c] = m;
    rstd[c] = rsqrtf(var + EPSBN);
}

// ---------------- normalize + relu + bf16 pack ----------------
// from NCHW x -> token-major bf16 dst[t*ldd + coff + c]
__global__ void k_norm_relu_nchw(const float* __restrict__ x, const float* __restrict__ mean,
                                 const float* __restrict__ rstd, const float* __restrict__ g,
                                 const float* __restrict__ beta, unsigned short* __restrict__ dst,
                                 int ldd, int coff) {
    for (int i = blockIdx.x * blockDim.x + threadIdx.x; i < NTOT; i += gridDim.x * blockDim.x) {
        const int b = i / (CCH * HWN);
        const int rem = i % (CCH * HWN);
        const int c = rem / HWN, hw = rem % HWN;
        float v = (x[i] - mean[c]) * rstd[c] * g[coff + c] + beta[coff + c];
        v = fmaxf(v, 0.f);
        dst[(size_t)(b * HWN + hw) * ldd + coff + c] = f2bf(v);
    }
}

// from token-major f32 src -> token-major bf16 dst
__global__ void k_norm_relu_tok(const float* __restrict__ src, const float* __restrict__ mean,
                                const float* __restrict__ rstd, const float* __restrict__ g,
                                const float* __restrict__ beta, unsigned short* __restrict__ dst,
                                int ldd, int coff) {
    const int total = TOK * CCH;
    for (int i = blockIdx.x * blockDim.x + threadIdx.x; i < total; i += gridDim.x * blockDim.x) {
        const int t = i / CCH, c = i % CCH;
        float v = (src[i] - mean[c]) * rstd[c] * g[coff + c] + beta[coff + c];
        v = fmaxf(v, 0.f);
        dst[(size_t)t * ldd + coff + c] = f2bf(v);
    }
}

// ---------------- WMMA bf16 GEMM (double-buffered, 32x32 per wave) ----------
// out[M=TOK, N=512] = A[M,K](bf16, row stride lda) * W[N,K]^T + bias.
// Block tile 128x64, BK=32, 256 threads = 8 waves.
// Wave grid 4(M) x 2(N): each wave owns 32x32 output = 2x2 WMMA tiles.
#define WMMA_BF16(a, b, c) \
    __builtin_amdgcn_wmma_f32_16x16x32_bf16(false, (a), false, (b), (short)0, (c), false, false)

__global__ __launch_bounds__(256, 2) void k_gemm_bf16(const unsigned short* __restrict__ A, int lda,
                                                      int K, const unsigned short* __restrict__ W,
                                                      const float* __restrict__ bias,
                                                      float* __restrict__ out, int outNCHW) {
    __shared__ __align__(16) unsigned short As[2][128 * 32];   // 2 x 8 KB
    __shared__ __align__(16) unsigned short Bs[2][64 * 32];    // 2 x 4 KB

    const int tid  = threadIdx.x;
    const int m0   = blockIdx.y * 128;
    const int n0   = blockIdx.x * 64;
    const int wave = tid >> 5, lane = tid & 31;
    const int hf   = lane >> 4;          // lane half
    const int r    = lane & 15;
    const int mg   = wave >> 1;          // 0..3 -> rows [mg*32, mg*32+32)
    const int ng   = wave & 1;           // 0..1 -> cols [ng*32, ng*32+32)

    // staging: A = 512 uint4 chunks (2/thread), B = 256 chunks (1/thread)
    const int srow = tid >> 2;           // 0..63
    const int scol = (tid & 3) * 8;      // 0,8,16,24
    const unsigned short* Ag0 = A + (size_t)(m0 + srow) * lda + scol;
    const unsigned short* Ag1 = A + (size_t)(m0 + srow + 64) * lda + scol;
    const unsigned short* Wg  = W + (size_t)(n0 + srow) * K + scol;

    v8f acc00 = {}, acc01 = {}, acc10 = {}, acc11 = {};

    // prologue: fill stage 0
    {
        uint4 a0 = *(const uint4*)Ag0;
        uint4 a1 = *(const uint4*)Ag1;
        uint4 b0 = *(const uint4*)Wg;
        *(uint4*)&As[0][srow * 32 + scol]        = a0;
        *(uint4*)&As[0][(srow + 64) * 32 + scol] = a1;
        *(uint4*)&Bs[0][srow * 32 + scol]        = b0;
    }
    __syncthreads();

    const int nsteps = K >> 5;
    const int mrow0 = (mg * 2 + 0) * 16 + r;
    const int mrow1 = (mg * 2 + 1) * 16 + r;
    const int nrow0 = (ng * 2 + 0) * 16 + r;
    const int nrow1 = (ng * 2 + 1) * 16 + r;
    int cur = 0;

    for (int s = 0; s < nsteps; ++s) {
        // prefetch next K-chunk into registers (uniform branch)
        uint4 pa0, pa1, pb0;
        const bool has_next = (s + 1) < nsteps;
        if (has_next) {
            const int k1 = (s + 1) << 5;
            pa0 = *(const uint4*)(Ag0 + k1);
            pa1 = *(const uint4*)(Ag1 + k1);
            pb0 = *(const uint4*)(Wg + k1);
        }

        const unsigned short* as = As[cur];
        const unsigned short* bs = Bs[cur];
        // A fragment (16-bit A 16x32): VGPR pairs 0-3 <- K=[hf*8,+8), 4-7 <- K=[16+hf*8,+8)
        union { uint4 q[2]; v16bf v; } fa0, fa1, fb0, fb1;
        fa0.q[0] = *(const uint4*)&as[mrow0 * 32 + hf * 8];
        fa0.q[1] = *(const uint4*)&as[mrow0 * 32 + 16 + hf * 8];
        fa1.q[0] = *(const uint4*)&as[mrow1 * 32 + hf * 8];
        fa1.q[1] = *(const uint4*)&as[mrow1 * 32 + 16 + hf * 8];
        // B fragment (16-bit B 32x16): lane = column N, same K order as A: [hf*8,+8), [16+hf*8,+8)
        fb0.q[0] = *(const uint4*)&bs[nrow0 * 32 + hf * 8];
        fb0.q[1] = *(const uint4*)&bs[nrow0 * 32 + 16 + hf * 8];
        fb1.q[0] = *(const uint4*)&bs[nrow1 * 32 + hf * 8];
        fb1.q[1] = *(const uint4*)&bs[nrow1 * 32 + 16 + hf * 8];

        acc00 = WMMA_BF16(fa0.v, fb0.v, acc00);
        acc01 = WMMA_BF16(fa0.v, fb1.v, acc01);
        acc10 = WMMA_BF16(fa1.v, fb0.v, acc10);
        acc11 = WMMA_BF16(fa1.v, fb1.v, acc11);

        if (has_next) {
            const int nxt = cur ^ 1;
            *(uint4*)&As[nxt][srow * 32 + scol]        = pa0;
            *(uint4*)&As[nxt][(srow + 64) * 32 + scol] = pa1;
            *(uint4*)&Bs[nxt][srow * 32 + scol]        = pb0;
        }
        __syncthreads();     // single barrier per K-step (double buffered)
        cur ^= 1;
    }

    // epilogue: bias + store (optionally scatter straight to NCHW)
    const int nbase = n0 + ng * 32 + r;
    #pragma unroll
    for (int j = 0; j < 2; ++j) {
        const int n = nbase + j * 16;
        const float bv = bias[n];
        #pragma unroll
        for (int i = 0; i < 2; ++i) {
            const v8f acc = (i == 0) ? (j == 0 ? acc00 : acc01)
                                     : (j == 0 ? acc10 : acc11);
            #pragma unroll
            for (int e = 0; e < 8; ++e) {
                const int m = m0 + mg * 32 + i * 16 + hf * 8 + e;
                if (!outNCHW)
                    out[(size_t)m * CCH + n] = acc[e] + bv;
                else
                    out[((size_t)(m / HWN) * CCH + n) * HWN + (m % HWN)] = acc[e] + bv;
            }
        }
    }
}

// ---------------- MHSA attention core (tiny: 9x9 per head) ----------------
// grid = (HEADS=4, B=4096), 128 threads = 128 d-channels
__global__ __launch_bounds__(128) void k_attention(const float* __restrict__ qb,
                                                   const float* __restrict__ kb,
                                                   const float* __restrict__ vb,
                                                   const float* __restrict__ relh,
                                                   const float* __restrict__ relw,
                                                   float* __restrict__ out) {
    __shared__ float qs[128 * 9], ks[128 * 9], ps[128 * 9], at[81];
    const int h = blockIdx.x, b = blockIdx.y, t = threadIdx.x;
    const size_t base = (size_t)b * HWN * CCH + h * 128 + t;

    float vv[9];
    #pragma unroll
    for (int m = 0; m < 9; ++m) {
        qs[t * 9 + m] = qb[base + (size_t)m * CCH];
        ks[t * 9 + m] = kb[base + (size_t)m * CCH];
        vv[m]         = vb[base + (size_t)m * CCH];
        // pos[h,d,j] = rel_h[h,d,j%3] + rel_w[h,d,j/3]
        ps[t * 9 + m] = relh[(h * 128 + t) * 3 + (m % 3)] + relw[(h * 128 + t) * 3 + (m / 3)];
    }
    __syncthreads();

    if (t < 81) {                        // scores: cc + cp
        const int n = t / 9, m = t % 9;
        float s = 0.f;
        for (int d = 0; d < 128; ++d)
            s += qs[d * 9 + n] * ks[d * 9 + m] + ps[d * 9 + n] * qs[d * 9 + m];
        at[t] = s;
    }
    __syncthreads();

    if (t < 9) {                         // softmax over m
        float mx = at[t * 9];
        #pragma unroll
        for (int m = 1; m < 9; ++m) mx = fmaxf(mx, at[t * 9 + m]);
        float e[9], sm = 0.f;
        #pragma unroll
        for (int m = 0; m < 9; ++m) { e[m] = __expf(at[t * 9 + m] - mx); sm += e[m]; }
        const float inv = 1.f / sm;
        #pragma unroll
        for (int m = 0; m < 9; ++m) at[t * 9 + m] = e[m] * inv;
    }
    __syncthreads();

    #pragma unroll
    for (int n = 0; n < 9; ++n) {        // out[d,n] = sum_m v[d,m] attn[n,m]
        float o = 0.f;
        #pragma unroll
        for (int m = 0; m < 9; ++m) o += vv[m] * at[n * 9 + m];
        out[base + (size_t)n * CCH] = o;
    }
}

// ---------------------------------------------------------------------------
extern "C" void kernel_launch(void* const* d_in, const int* in_sizes, int n_in,
                              void* d_out, int out_size, void* d_ws, size_t ws_size,
                              hipStream_t stream) {
    (void)in_sizes; (void)n_in; (void)out_size; (void)ws_size;
    const float* x     = (const float*)d_in[0];
    const float* bn1_g = (const float*)d_in[1];
    const float* bn1_b = (const float*)d_in[2];
    const float* w1    = (const float*)d_in[3];
    const float* b1    = (const float*)d_in[4];
    const float* q_w   = (const float*)d_in[5];
    const float* q_b   = (const float*)d_in[6];
    const float* k_w   = (const float*)d_in[7];
    const float* k_b   = (const float*)d_in[8];
    const float* v_w   = (const float*)d_in[9];
    const float* v_b   = (const float*)d_in[10];
    const float* rel_h = (const float*)d_in[11];
    const float* rel_w = (const float*)d_in[12];
    const float* bn2_g = (const float*)d_in[13];
    const float* bn2_b = (const float*)d_in[14];
    const float* w2    = (const float*)d_in[15];
    const float* b2    = (const float*)d_in[16];
    const float* bn3_g = (const float*)d_in[17];
    const float* bn3_b = (const float*)d_in[18];
    const float* w3    = (const float*)d_in[19];
    const float* b3    = (const float*)d_in[20];
    float* out = (float*)d_out;

    // -------- workspace carving --------
    char* ws = (char*)d_ws;
    size_t off = 0;
    auto alloc = [&](size_t bytes) -> void* {
        void* p = ws + off;
        off = (off + bytes + 255) & ~(size_t)255;
        return p;
    };
    unsigned short* w1b  = (unsigned short*)alloc((size_t)512 * 512 * 2);
    unsigned short* qwb  = (unsigned short*)alloc((size_t)512 * 512 * 2);
    unsigned short* kwb  = (unsigned short*)alloc((size_t)512 * 512 * 2);
    unsigned short* vwb  = (unsigned short*)alloc((size_t)512 * 512 * 2);
    unsigned short* w2b  = (unsigned short*)alloc((size_t)512 * 1024 * 2);
    unsigned short* w3b  = (unsigned short*)alloc((size_t)512 * 1536 * 2);
    float* mean1  = (float*)alloc(512 * 4);
    float* rstd1  = (float*)alloc(512 * 4);
    float* meanA1 = (float*)alloc(512 * 4);
    float* rstdA1 = (float*)alloc(512 * 4);
    float* meanA2 = (float*)alloc(512 * 4);
    float* rstdA2 = (float*)alloc(512 * 4);
    float* partial = (float*)alloc((size_t)144 * 1024 * 4);
    unsigned short* bufA = (unsigned short*)alloc((size_t)TOK * 1536 * 2); // concat GEMM input
    unsigned short* actb = (unsigned short*)alloc((size_t)TOK * 512 * 2);  // bf16 of o3/o7
    float* P1   = (float*)alloc((size_t)TOK * 512 * 4);  // o3 / o7
    float* P2   = (float*)alloc((size_t)TOK * 512 * 4);  // q
    float* P3   = (float*)alloc((size_t)TOK * 512 * 4);  // k
    float* P4   = (float*)alloc((size_t)TOK * 512 * 4);  // v
    float* ATT1 = (float*)alloc((size_t)TOK * 512 * 4);  // o3_att
    float* ATT2 = (float*)alloc((size_t)TOK * 512 * 4);  // o7_att

    const float invN = 1.0f / (float)TOK;
    const dim3 gemmGrid(CCH / 64, TOK / 128);   // (8, 288)
    const dim3 attGrid(4, NB);

    // -------- weights -> bf16 --------
    k_f2bf<<<256, 256, 0, stream>>>(w1,  w1b, 512 * 512);
    k_f2bf<<<256, 256, 0, stream>>>(q_w, qwb, 512 * 512);
    k_f2bf<<<256, 256, 0, stream>>>(k_w, kwb, 512 * 512);
    k_f2bf<<<256, 256, 0, stream>>>(v_w, vwb, 512 * 512);
    k_f2bf<<<256, 256, 0, stream>>>(w2,  w2b, 512 * 1024);
    k_f2bf<<<512, 256, 0, stream>>>(w3,  w3b, 512 * 1536);

    // -------- stage 1: bn1 -> relu -> w1 -> o3 --------
    k_stats_nchw_partial<<<128, 256, 0, stream>>>(x, partial);
    k_stats_finalize<<<1, 512, 0, stream>>>(partial, 128, invN, mean1, rstd1);
    k_norm_relu_nchw<<<2048, 256, 0, stream>>>(x, mean1, rstd1, bn1_g, bn1_b, bufA, 512, 0);
    k_gemm_bf16<<<gemmGrid, 256, 0, stream>>>(bufA, 512, 512, w1b, b1, P1, 0);

    // -------- MHSA(o3) --------
    k_f2bf<<<4096, 256, 0, stream>>>(P1, actb, TOK * 512);
    k_gemm_bf16<<<gemmGrid, 256, 0, stream>>>(actb, 512, 512, qwb, q_b, P2, 0);
    k_gemm_bf16<<<gemmGrid, 256, 0, stream>>>(actb, 512, 512, kwb, k_b, P3, 0);
    k_gemm_bf16<<<gemmGrid, 256, 0, stream>>>(actb, 512, 512, vwb, v_b, P4, 0);
    k_attention<<<attGrid, 128, 0, stream>>>(P2, P3, P4, rel_h, rel_w, ATT1);

    // -------- stage 2: bn2(concat(x, o3_att)) -> relu -> w2 -> o7 --------
    k_stats_tok_partial<<<144, 256, 0, stream>>>(ATT1, partial);
    k_stats_finalize<<<1, 512, 0, stream>>>(partial, 144, invN, meanA1, rstdA1);
    k_norm_relu_nchw<<<2048, 256, 0, stream>>>(x, mean1, rstd1, bn2_g, bn2_b, bufA, 1024, 0);
    k_norm_relu_tok<<<2048, 256, 0, stream>>>(ATT1, meanA1, rstdA1, bn2_g, bn2_b, bufA, 1024, 512);
    k_gemm_bf16<<<gemmGrid, 256, 0, stream>>>(bufA, 1024, 1024, w2b, b2, P1, 0);

    // -------- MHSA(o7), same projection weights --------
    k_f2bf<<<4096, 256, 0, stream>>>(P1, actb, TOK * 512);
    k_gemm_bf16<<<gemmGrid, 256, 0, stream>>>(actb, 512, 512, qwb, q_b, P2, 0);
    k_gemm_bf16<<<gemmGrid, 256, 0, stream>>>(actb, 512, 512, kwb, k_b, P3, 0);
    k_gemm_bf16<<<gemmGrid, 256, 0, stream>>>(actb, 512, 512, vwb, v_b, P4, 0);
    k_attention<<<attGrid, 128, 0, stream>>>(P2, P3, P4, rel_h, rel_w, ATT2);

    // -------- stage 3: bn3(concat(x, o3_att, o7_att)) -> relu -> w3 -> out (NCHW) --------
    k_stats_tok_partial<<<144, 256, 0, stream>>>(ATT2, partial);
    k_stats_finalize<<<1, 512, 0, stream>>>(partial, 144, invN, meanA2, rstdA2);
    k_norm_relu_nchw<<<2048, 256, 0, stream>>>(x, mean1, rstd1, bn3_g, bn3_b, bufA, 1536, 0);
    k_norm_relu_tok<<<2048, 256, 0, stream>>>(ATT1, meanA1, rstdA1, bn3_g, bn3_b, bufA, 1536, 512);
    k_norm_relu_tok<<<2048, 256, 0, stream>>>(ATT2, meanA2, rstdA2, bn3_g, bn3_b, bufA, 1536, 1024);
    k_gemm_bf16<<<gemmGrid, 256, 0, stream>>>(bufA, 1536, 1536, w3b, b3, out, 1);
}
